// E3DLSTM_Model_18846316495147
// MI455X (gfx1250) — matrix operand order repaired and checked by reading, not compiled, not run on hardware
//
#include <hip/hip_runtime.h>

// ---------------------------------------------------------------------------
// E3D-LSTM on MI455X (gfx1250): conv3d as implicit GEMM via v_wmma_f32_16x16x32_bf16
// with async global->LDS im2col staging (ASYNCcnt) and LDS double buffering.
// ---------------------------------------------------------------------------

typedef __attribute__((ext_vector_type(16))) __bf16 v16bf;
typedef __attribute__((ext_vector_type(8)))  float  v8f;

#define HW    4096     // 64*64
#define DHW   16384    // 4*64*64
#define CBUF  524288   // 32*DHW (one batch's (C,D,H,W))
#define BUF   2097152  // 4*CBUF (one full (B,C,D,H,W) tensor)
#define MTOT  43200    // B(4)*od(3)*3600 output positions of VALID conv
#define SPAT  3600     // 60*60

union FragAB { v16bf v; __bf16 h[16]; };

// Implicit-GEMM conv3d (VALID, kernel 2x5x5), output (4,32,3,60,60) into scr.
// A: 16(M)x32(K) bf16 per wave, B: 32(K)x16(N) (two N halves), C: 16x16 f32.
// Block = 128 threads = 4 waves; K-loop double-buffered: async im2col of tile
// kt+1 into LDS overlaps fragment packing + WMMA of tile kt.
__global__ __launch_bounds__(128) void conv_wmma(
    const float* __restrict__ in, const float* __restrict__ wgt,
    float* __restrict__ scr, int Cin, int K) {
  __shared__ float As[2][64 * 33];   // 64 M-rows x 32 K (f32, stride 33)
  __shared__ float Bs[2][32 * 33];   // 32 N x 32 K (N-major; K-pairs contiguous)
  const int tid  = threadIdx.x;
  const int lane = tid & 31, wv = tid >> 5;
  const int hs   = lane >> 4, l16 = lane & 15;
  const int mBase = blockIdx.x * 64;
  const int kk   = tid & 31;   // staged K index = lane within staging group
  const int grp  = tid >> 5;   // 0..3

  // k-invariant part of the im2col address for the 16 rows this thread stages
  // (rows m = mBase+grp+4i, consecutive stride 4 -> incremental carry walk,
  //  one unsigned division chain instead of 16)
  int aBase[16];
  {
    unsigned m0 = (unsigned)(mBase + grp);
    unsigned bb = m0 / 10800u;   unsigned rem = m0 - bb * 10800u;
    unsigned od = rem / 3600u;   unsigned p   = rem - od * 3600u;
    unsigned oh = p / 60u;       unsigned ow  = p - oh * 60u;
    int base = (int)(bb * (unsigned)(Cin * 4 * HW) + od * HW + oh * 64 + ow);
#pragma unroll
    for (int i = 0; i < 16; ++i) {
      aBase[i] = base;
      ow += 4; base += 4;
      if (ow >= 60) {
        ow -= 60; oh += 1; base += 4;          // 64 - 60
        if (oh >= 60) {
          oh -= 60; od += 1; base += 256;      // HW - 60*64
          if (od >= 3) { od -= 3; base += Cin * 4 * HW - 3 * HW; }
        }
      }
    }
  }

  const int nK = (K + 31) >> 5;

  // async im2col: per-lane gather straight into LDS (no VGPR landing)
  auto stage = [&](int kt, int buf) {
    int k = (kt << 5) + kk;
    if (k < K) {
      int kw = k % 5, kh = (k / 5) % 5, kd = (k / 25) & 1, ci = k / 50;
      int kOff = ci * 4 * HW + kd * HW + kh * 64 + kw;
#pragma unroll
      for (int i = 0; i < 16; ++i) {
        unsigned lds = (unsigned)(uintptr_t)&As[buf][(grp + (i << 2)) * 33 + kk];
        unsigned long long ga =
            (unsigned long long)(uintptr_t)&in[aBase[i] + kOff];
        asm volatile("global_load_async_to_lds_b32 %0, %1, off"
                     :: "v"(lds), "v"(ga) : "memory");
      }
#pragma unroll
      for (int i = 0; i < 8; ++i) {
        int n = grp + (i << 2);
        unsigned lds = (unsigned)(uintptr_t)&Bs[buf][n * 33 + kk];
        unsigned long long ga =
            (unsigned long long)(uintptr_t)&wgt[n * K + k];
        asm volatile("global_load_async_to_lds_b32 %0, %1, off"
                     :: "v"(lds), "v"(ga) : "memory");
      }
    } else {  // zero-fill K tail (K=400 case) with plain DS stores
#pragma unroll
      for (int i = 0; i < 16; ++i) As[buf][(grp + (i << 2)) * 33 + kk] = 0.f;
#pragma unroll
      for (int i = 0; i < 8; ++i)  Bs[buf][(grp + (i << 2)) * 33 + kk] = 0.f;
    }
  };

  v8f acc0 = {}; v8f acc1 = {};
  int buf = 0;

  stage(0, 0);
  asm volatile("s_wait_asynccnt 0x0" ::: "memory");
  __syncthreads();

  for (int kt = 0; kt < nK; ++kt) {
    if (kt + 1 < nK) stage(kt + 1, buf ^ 1);   // overlap with compute below

    // ---- pack bf16 fragments per ISA 7.12.2 (16-bit A 16x32; B mirrored)
    // native __bf16 casts -> hardware v_cvt (RNE), no integer-rounding VALU
    FragAB a, b0, b1;
    const float* Ar = &As[buf][(wv * 16 + l16) * 33];
#pragma unroll
    for (int j = 0; j < 8; ++j) {
      int kb = (j < 4) ? (j * 2 + hs * 8) : (16 + (j - 4) * 2 + hs * 8);
      a.h[2 * j]     = (__bf16)Ar[kb];
      a.h[2 * j + 1] = (__bf16)Ar[kb + 1];
    }
    const float* Br0 = &Bs[buf][l16 * 33];
    const float* Br1 = &Bs[buf][(16 + l16) * 33];
#pragma unroll
    for (int j = 0; j < 8; ++j) {
      int kb = (j < 4) ? (j * 2 + hs * 8) : (16 + (j - 4) * 2 + hs * 8);
      b0.h[2 * j]     = (__bf16)Br0[kb];
      b0.h[2 * j + 1] = (__bf16)Br0[kb + 1];
      b1.h[2 * j]     = (__bf16)Br1[kb];
      b1.h[2 * j + 1] = (__bf16)Br1[kb + 1];
    }
    acc0 = __builtin_amdgcn_wmma_f32_16x16x32_bf16(false, a.v, false, b0.v,
                                                   (short)0, acc0, false, false);
    acc1 = __builtin_amdgcn_wmma_f32_16x16x32_bf16(false, a.v, false, b1.v,
                                                   (short)0, acc1, false, false);

    asm volatile("s_wait_asynccnt 0x0" ::: "memory");
    __syncthreads();
    buf ^= 1;
  }

  // ---- epilogue: C/D layout (lane=N, vgpr r -> M=r+8*hs) -> (b,32,3,60,60)
  {
    unsigned m0 = (unsigned)(mBase + wv * 16 + hs * 8);
    unsigned bb = m0 / 10800u;  unsigned rem = m0 - bb * 10800u;
    unsigned od = rem / 3600u;  unsigned p   = rem - od * 3600u;
#pragma unroll
    for (int r = 0; r < 8; ++r) {
      scr[((bb * 32 + l16) * 3 + od) * SPAT + p]        = acc0[r];
      scr[((bb * 32 + 16 + l16) * 3 + od) * SPAT + p]   = acc1[r];
      p += 1;
      if (p >= 3600u) { p = 0; od += 1; if (od >= 3u) { od = 0; bb += 1; } }
    }
  }
}

// nearest-resize (3,60,60)->(4,64,64), optional bias, set or accumulate.
__global__ void resize_accum(const float* __restrict__ scr,
                             const float* __restrict__ bias,
                             float* __restrict__ dst, int add) {
  int idx = blockIdx.x * 256 + threadIdx.x;
  int wz = idx & 63, hz = (idx >> 6) & 63, dz = (idx >> 12) & 3;
  int cc = (idx >> 14) & 31, b = idx >> 19;
  int od = (dz * 3) >> 2, oh = (hz * 60) >> 6, ow = (wz * 60) >> 6;
  float v = scr[((b * 32 + cc) * 3 + od) * SPAT + oh * 60 + ow];
  if (bias) v += bias[cc];
  dst[idx] = add ? (dst[idx] + v) : v;
}

// _ln over (D,H,W) per (b,c), then sigmoid (act=0) or tanh (act=1), in place.
__global__ __launch_bounds__(256) void ln_act(float* __restrict__ t, int act) {
  __shared__ float rs[256], rss[256];
  float* p = t + (size_t)blockIdx.x * DHW;
  float s = 0.f, ss = 0.f;
  for (int i = threadIdx.x; i < DHW; i += 256) { float v = p[i]; s += v; ss += v * v; }
  rs[threadIdx.x] = s; rss[threadIdx.x] = ss;
  __syncthreads();
  for (int o = 128; o > 0; o >>= 1) {
    if (threadIdx.x < o) { rs[threadIdx.x] += rs[threadIdx.x + o];
                           rss[threadIdx.x] += rss[threadIdx.x + o]; }
    __syncthreads();
  }
  float mu = rs[0] * (1.f / DHW);
  float rstd = rsqrtf(rss[0] * (1.f / DHW) - mu * mu + 1e-5f);
  for (int i = threadIdx.x; i < DHW; i += 256) {
    float v = (p[i] - mu) * rstd;
    p[i] = act ? tanhf(v) : 1.f / (1.f + __expf(-v));
  }
}

// scores[b,l] = <r[b], c_hist[l][b]> / 128
__global__ __launch_bounds__(256) void score_k(const float* __restrict__ r,
    const float* __restrict__ c0, const float* __restrict__ c1,
    float* __restrict__ small) {
  __shared__ float red[256];
  int b = blockIdx.x & 3, l = blockIdx.x >> 2;
  const float* ch = (l ? c1 : c0) + (size_t)b * CBUF;
  const float* rb = r + (size_t)b * CBUF;
  float s = 0.f;
  for (int i = threadIdx.x; i < CBUF; i += 256) s += rb[i] * ch[i];
  red[threadIdx.x] = s; __syncthreads();
  for (int o = 128; o > 0; o >>= 1) {
    if (threadIdx.x < o) red[threadIdx.x] += red[threadIdx.x + o];
    __syncthreads();
  }
  if (threadIdx.x == 0) small[b * 2 + l] = red[0] * (1.0f / 128.0f);
}

// recall = sum_l softmax_over_b(scores)[b,l] * c_hist[l][b]  (softmax axis=0!)
__global__ void recall_k(const float* __restrict__ small,
    const float* __restrict__ c0, const float* __restrict__ c1,
    float* __restrict__ rec) {
  int idx = blockIdx.x * 256 + threadIdx.x;
  int b = idx >> 19;
  float m0 = -1e30f, m1 = -1e30f;
  for (int bb = 0; bb < 4; ++bb) { m0 = fmaxf(m0, small[bb * 2]);
                                   m1 = fmaxf(m1, small[bb * 2 + 1]); }
  float s0 = 0.f, s1 = 0.f;
  for (int bb = 0; bb < 4; ++bb) { s0 += __expf(small[bb * 2] - m0);
                                   s1 += __expf(small[bb * 2 + 1] - m1); }
  float a0 = __expf(small[b * 2] - m0) / s0;
  float a1 = __expf(small[b * 2 + 1] - m1) / s1;
  rec[idx] = a0 * c0[idx] + a1 * c1[idx];
}

// _ln4 stats of (c_hist[-1] + recall) per b over (C,D,H,W)
__global__ __launch_bounds__(256) void cstat_k(const float* __restrict__ c1,
    const float* __restrict__ rec, float* __restrict__ small) {
  __shared__ float rs[256], rss[256];
  int b = blockIdx.x;
  const float* pa = c1 + (size_t)b * CBUF;
  const float* pb = rec + (size_t)b * CBUF;
  float s = 0.f, ss = 0.f;
  for (int i = threadIdx.x; i < CBUF; i += 256) { float v = pa[i] + pb[i]; s += v; ss += v * v; }
  rs[threadIdx.x] = s; rss[threadIdx.x] = ss;
  __syncthreads();
  for (int o = 128; o > 0; o >>= 1) {
    if (threadIdx.x < o) { rs[threadIdx.x] += rs[threadIdx.x + o];
                           rss[threadIdx.x] += rss[threadIdx.x + o]; }
    __syncthreads();
  }
  if (threadIdx.x == 0) {
    float mu = rs[0] * (1.f / CBUF);
    small[8 + b]  = mu;
    small[12 + b] = rsqrtf(rss[0] * (1.f / CBUF) - mu * mu + 1e-5f);
  }
}

// c = i*g + ln4(c_hist[-1] + recall) with affine (lw, lb broadcast over b)
__global__ void cnew_k(const float* __restrict__ gi, const float* __restrict__ gg,
                       const float* __restrict__ c1, const float* __restrict__ rec,
                       const float* __restrict__ small,
                       const float* __restrict__ lw, const float* __restrict__ lb,
                       float* __restrict__ cn) {
  int idx = blockIdx.x * 256 + threadIdx.x;
  int b = idx >> 19; int j = idx & (CBUF - 1);
  float x = c1[idx] + rec[idx];
  cn[idx] = gi[idx] * gg[idx] + (x - small[8 + b]) * small[12 + b] * lw[j] + lb[j];
}

// m2 = i'*g' + f'*m
__global__ void mnew_k(const float* __restrict__ ip, const float* __restrict__ gp,
                       const float* __restrict__ fp, const float* __restrict__ m,
                       float* __restrict__ m2) {
  int idx = blockIdx.x * 256 + threadIdx.x;
  m2[idx] = ip[idx] * gp[idx] + fp[idx] * m[idx];
}

// h = o * tanh(conv1x1x1(cat(c, m2)));  also scatter into d_out at step t.
__global__ void hnew_k(const float* __restrict__ o, const float* __restrict__ c,
                       const float* __restrict__ m2, const float* __restrict__ w111,
                       const float* __restrict__ b111,
                       float* __restrict__ h2, float* __restrict__ out, int t) {
  int idx = blockIdx.x * 256 + threadIdx.x;
  int pos = idx & (DHW - 1); int co = (idx >> 14) & 31; int b = idx >> 19;
  float acc = b111[co];
  const float* cb = c  + (size_t)b * CBUF + pos;
  const float* mb = m2 + (size_t)b * CBUF + pos;
  const float* w  = w111 + co * 64;
#pragma unroll
  for (int ci = 0; ci < 32; ++ci) acc += w[ci] * cb[(size_t)ci * DHW];
#pragma unroll
  for (int ci = 0; ci < 32; ++ci) acc += w[32 + ci] * mb[(size_t)ci * DHW];
  float h = o[idx] * tanhf(acc);
  h2[idx] = h;
  out[(((size_t)b * 128 + t * 32 + co) << 14) + pos] = h;
}

__global__ void fill0(float* __restrict__ p) {
  p[blockIdx.x * 256 + threadIdx.x] = 0.f;
}

// ---------------------------------------------------------------------------
extern "C" void kernel_launch(void* const* d_in, const int* in_sizes, int n_in,
                              void* d_out, int out_size, void* d_ws, size_t ws_size,
                              hipStream_t stream) {
  (void)in_sizes; (void)n_in; (void)out_size; (void)ws_size;
  const float* x    = (const float*)d_in[0];   // (4,4,8,4,64,64)
  const float* Wx   = (const float*)d_in[1];   // (7,32,8,2,5,5)
  const float* bx   = (const float*)d_in[2];   // (7,32)
  const float* Wh   = (const float*)d_in[3];   // (9,32,32,2,5,5)
  const float* w111 = (const float*)d_in[4];   // (32,64,1,1,1)
  const float* b111 = (const float*)d_in[5];   // (32)
  const float* lw   = (const float*)d_in[6];   // (32,4,64,64)
  const float* lb   = (const float*)d_in[7];
  float* out = (float*)d_out;

  float* W = (float*)d_ws;
  float* c0 = W;              float* c1 = W + (size_t)BUF;
  float* c2 = W + 2*(size_t)BUF;
  float* m0 = W + 3*(size_t)BUF;  float* m1 = W + 4*(size_t)BUF;
  float* h0 = W + 5*(size_t)BUF;  float* h1 = W + 6*(size_t)BUF;
  float* t0 = W + 7*(size_t)BUF;  float* t1 = W + 8*(size_t)BUF;
  float* t2 = W + 9*(size_t)BUF;
  float* rec = W + 10*(size_t)BUF;
  float* scr = W + 11*(size_t)BUF;   // conv VALID scratch (needs 1,382,400 floats)
  float* small = W + 12*(size_t)BUF; // scores[8], mu[4], rstd[4]

  dim3 B256(256), Gbuf(BUF / 256);
  fill0<<<Gbuf, B256, 0, stream>>>(c0);
  fill0<<<Gbuf, B256, 0, stream>>>(c1);
  fill0<<<Gbuf, B256, 0, stream>>>(m0);
  fill0<<<Gbuf, B256, 0, stream>>>(h0);

  const int KX = 400, KH = 1600;
  auto conv = [&](const float* in_, const float* w_, int Cin, int K) {
    conv_wmma<<<dim3(MTOT / 64), dim3(128), 0, stream>>>(in_, w_, scr, Cin, K);
  };
  auto res = [&](float* dst, const float* bias, int add) {
    resize_accum<<<Gbuf, B256, 0, stream>>>(scr, bias, dst, add);
  };

  float *hc0 = c0, *hc1 = c1, *cn = c2, *pm = m0, *pm2 = m1, *ph = h0, *ph2 = h1;
  for (int t = 0; t < 4; ++t) {
    const float* xt = x + (size_t)t * 4 * 8 * DHW;
    // r, i, g gates (use old h)
    conv(xt, Wx + 0 * 32 * KX, 8, KX);  res(t0, bx + 0 * 32, 0);
    conv(ph, Wh + 0 * 32 * KH, 32, KH); res(t0, nullptr, 1);
    conv(xt, Wx + 1 * 32 * KX, 8, KX);  res(t1, bx + 1 * 32, 0);
    conv(ph, Wh + 1 * 32 * KH, 32, KH); res(t1, nullptr, 1);
    conv(xt, Wx + 2 * 32 * KX, 8, KX);  res(t2, bx + 2 * 32, 0);
    conv(ph, Wh + 2 * 32 * KH, 32, KH); res(t2, nullptr, 1);
    ln_act<<<dim3(128), B256, 0, stream>>>(t0, 0);   // r = sigmoid
    ln_act<<<dim3(128), B256, 0, stream>>>(t1, 0);   // i = sigmoid
    ln_act<<<dim3(128), B256, 0, stream>>>(t2, 1);   // g = tanh
    // attention recall + new c
    score_k<<<dim3(8), B256, 0, stream>>>(t0, hc0, hc1, small);
    recall_k<<<Gbuf, B256, 0, stream>>>(small, hc0, hc1, rec);
    cstat_k<<<dim3(4), B256, 0, stream>>>(hc1, rec, small);
    cnew_k<<<Gbuf, B256, 0, stream>>>(t1, t2, hc1, rec, small, lw, lb, cn);
    // i', g', f' gates (use old m)
    conv(xt, Wx + 3 * 32 * KX, 8, KX);  res(t0, bx + 3 * 32, 0);
    conv(pm, Wh + 3 * 32 * KH, 32, KH); res(t0, nullptr, 1);
    conv(xt, Wx + 4 * 32 * KX, 8, KX);  res(t1, bx + 4 * 32, 0);
    conv(pm, Wh + 4 * 32 * KH, 32, KH); res(t1, nullptr, 1);
    conv(xt, Wx + 5 * 32 * KX, 8, KX);  res(t2, bx + 5 * 32, 0);
    conv(pm, Wh + 5 * 32 * KH, 32, KH); res(t2, nullptr, 1);
    ln_act<<<dim3(128), B256, 0, stream>>>(t0, 0);   // i' = sigmoid
    ln_act<<<dim3(128), B256, 0, stream>>>(t1, 1);   // g' = tanh
    ln_act<<<dim3(128), B256, 0, stream>>>(t2, 0);   // f' = sigmoid
    mnew_k<<<Gbuf, B256, 0, stream>>>(t0, t1, t2, pm, pm2);
    // o gate (old h, new c, new m2)
    conv(xt,  Wx + 6 * 32 * KX, 8, KX);  res(t0, bx + 6 * 32, 0);
    conv(ph,  Wh + 6 * 32 * KH, 32, KH); res(t0, nullptr, 1);
    conv(cn,  Wh + 7 * 32 * KH, 32, KH); res(t0, nullptr, 1);
    conv(pm2, Wh + 8 * 32 * KH, 32, KH); res(t0, nullptr, 1);
    ln_act<<<dim3(128), B256, 0, stream>>>(t0, 0);   // o = sigmoid
    hnew_k<<<Gbuf, B256, 0, stream>>>(t0, cn, pm2, w111, b111, ph2, out, t);
    // rotate state
    float* tmp = hc0; hc0 = hc1; hc1 = cn; cn = tmp;
    tmp = pm; pm = pm2; pm2 = tmp;
    tmp = ph; ph = ph2; ph2 = tmp;
  }
}
